// DecoderRNN_86689619902939
// MI455X (gfx1250) — compile-verified, matrix operand order checked by reading, not run on hardware
//
#include <hip/hip_runtime.h>

#define BB      64
#define TT      32
#define EMBED   512
#define HIDDEN  1024
#define G4      4096      // 4*HIDDEN
#define VOCAB   32000
#define BT      2048      // BB*TT
#define LOGITS_ELEMS (BT * VOCAB)   // 65,536,000

typedef __attribute__((ext_vector_type(16))) __bf16       v16bf;
typedef __attribute__((ext_vector_type(8)))  float        v8f;
typedef __attribute__((ext_vector_type(4)))  unsigned int v4u;
typedef __attribute__((ext_vector_type(4)))  unsigned int su4;
typedef __attribute__((ext_vector_type(8)))  unsigned int su8;

union FragU { v16bf v; v4u u[2]; };

// ---------------------------------------------------------------------------
// Fragment loader for V_WMMA_F32_16X16X32_BF16 (ISA 7.12.2, 16-bit 16x32 A):
//   lane<16 : row = lane,    elems[0..7]=K(k0+0..7),  elems[8..15]=K(k0+16..23)
//   lane>=16: row = lane-16, elems[0..7]=K(k0+8..15), elems[8..15]=K(k0+24..31)
// Serves A (row-major, K contiguous) and B^T stored as W[n*ld + k].
// Works for global or LDS pointers (addrspace inferred after inlining).
// ---------------------------------------------------------------------------
__device__ __forceinline__ v16bf load_frag(const __bf16* tile_base, int ld,
                                           int k0, int lane) {
  const int half = (lane >> 4) & 1;
  const int row  = lane & 15;
  FragU f;
  const __bf16* p = tile_base + row * ld + k0 + half * 8;
  f.u[0] = *(const v4u*)(p);       // K = k0 + half*8 .. +7
  f.u[1] = *(const v4u*)(p + 16);  // K = k0 + 16 + half*8 .. +7
  return f.v;
}

#define WMMA_BF16(a, b, c) \
  __builtin_amdgcn_wmma_f32_16x16x32_bf16(false, (a), false, (b), (short)0, (c), false, false)

// ---------------------------------------------------------------------------
// Tensor Data Mover: issue a 2D tile load (tile_h rows x tile_w bf16 elems,
// row stride = stride_elems) into LDS at byte offset lds_off.
// D# per ISA 8.3/8.4: group0 = {count=1, lds_addr, global_addr[56:0], type=2},
// group1 = {wg_mask=0, data_size=1(2B), tensor dims = tile dims, tile dims,
//           tensor_dim0_stride}.  Issued once per wave; tracked by TENSORcnt.
// ---------------------------------------------------------------------------
__device__ __forceinline__ void tdm_load_tile_2d(unsigned lds_off,
                                                 const __bf16* gptr,
                                                 unsigned stride_elems,
                                                 unsigned tile_w,
                                                 unsigned tile_h) {
  unsigned long long ga = (unsigned long long)(size_t)gptr;
  su4 g0;
  g0[0] = 0x1u;                                   // count=1, is_restore=0
  g0[1] = lds_off;                                // lds_addr (bytes)
  g0[2] = (unsigned)(ga & 0xffffffffu);           // global_addr[31:0]
  g0[3] = (unsigned)((ga >> 32) & 0x01ffffffu)    // global_addr[56:32]
          | (2u << 30);                           // type = 2 ("image")
  su8 g1;
  g1[0] = 0x1u << 16;                             // data_size=1 (2 bytes)
  g1[1] = (tile_w & 0xffffu) << 16;               // tensor_dim0[15:0]
  g1[2] = ((tile_w >> 16) & 0xffffu)              // tensor_dim0[31:16]
          | ((tile_h & 0xffffu) << 16);           // tensor_dim1[15:0]
  g1[3] = ((tile_h >> 16) & 0xffffu)              // tensor_dim1[31:16]
          | ((tile_w & 0xffffu) << 16);           // tile_dim0
  g1[4] = (tile_h & 0xffffu);                     // tile_dim1 (tile_dim2=0)
  g1[5] = stride_elems;                           // tensor_dim0_stride[31:0]
  g1[6] = 0u;
  g1[7] = 0u;
  asm volatile("tensor_load_to_lds %0, %1" : : "s"(g0), "s"(g1) : "memory");
}

__device__ __forceinline__ float sigf(float x) {
  return 1.0f / (1.0f + __expf(-x));
}

// ------------------------- element-wise prep kernels -----------------------

__global__ void cvt_bf16_contig(const float* __restrict__ src,
                                __bf16* __restrict__ dst, int n) {
  int i = blockIdx.x * blockDim.x + threadIdx.x;
  if (i < n) dst[i] = (__bf16)src[i];
}

__global__ void cvt_wword(const float* __restrict__ W_ih,
                          __bf16* __restrict__ dst) {
  int i = blockIdx.x * blockDim.x + threadIdx.x;       // G4*EMBED
  if (i >= G4 * EMBED) return;
  int r = i >> 9;
  int c = i & (EMBED - 1);
  dst[i] = (__bf16)W_ih[r * (2 * EMBED) + c];
}

__global__ void embed_gather(const int* __restrict__ tokens,
                             const float* __restrict__ table,
                             __bf16* __restrict__ out) {
  int i = blockIdx.x * blockDim.x + threadIdx.x;       // BT*EMBED
  if (i >= BT * EMBED) return;
  int m = i >> 9;
  int e = i & (EMBED - 1);
  int tok = tokens[m];
  out[i] = (__bf16)table[tok * EMBED + e];
}

__global__ void enc_proj_kernel(const float* __restrict__ enc,
                                const float* __restrict__ W_ih,
                                const float* __restrict__ b_ih,
                                const float* __restrict__ b_hh,
                                float* __restrict__ enc_proj) {
  int i = blockIdx.x * blockDim.x + threadIdx.x;       // BB*G4
  if (i >= BB * G4) return;
  int b = i >> 12;
  int g = i & (G4 - 1);
  const float* w = W_ih + g * (2 * EMBED) + EMBED;
  const float* x = enc + b * EMBED;
  float acc = b_ih[g] + b_hh[g];
#pragma unroll 8
  for (int k = 0; k < EMBED; ++k) acc += x[k] * w[k];
  enc_proj[i] = acc;
}

__global__ void init_state(const float* __restrict__ h0,
                           const float* __restrict__ c0,
                           float* __restrict__ h_cur,
                           float* __restrict__ c_cur,
                           __bf16* __restrict__ h_bf) {
  int i = blockIdx.x * blockDim.x + threadIdx.x;       // BB*HIDDEN
  if (i >= BB * HIDDEN) return;
  float h = h0[i];
  h_cur[i] = h;
  c_cur[i] = c0[i];
  h_bf[i]  = (__bf16)h;
}

// ------------------- register-blocked 32x64 WMMA tile body -----------------
// One wave computes a 32(M) x 64(N) C tile: 2 A-frags x 4 B-frags, 8 accums.
// acc[mi*4 + ni].  A row-major (lda), B^T row-major as W[n*ldb + k].
__device__ __forceinline__ void mma_32x64(const __bf16* A0, int lda,
                                          const __bf16* B0, int ldb,
                                          int K, int lane, v8f acc[8]) {
  for (int k0 = 0; k0 < K; k0 += 32) {
    v16bf a0 = load_frag(A0,            lda, k0, lane);
    v16bf a1 = load_frag(A0 + 16 * lda, lda, k0, lane);
    v16bf b0 = load_frag(B0,            ldb, k0, lane);
    v16bf b1 = load_frag(B0 + 16 * ldb, ldb, k0, lane);
    v16bf b2 = load_frag(B0 + 32 * ldb, ldb, k0, lane);
    v16bf b3 = load_frag(B0 + 48 * ldb, ldb, k0, lane);
    acc[0] = WMMA_BF16(a0, b0, acc[0]);
    acc[1] = WMMA_BF16(a0, b1, acc[1]);
    acc[2] = WMMA_BF16(a0, b2, acc[2]);
    acc[3] = WMMA_BF16(a0, b3, acc[3]);
    acc[4] = WMMA_BF16(a1, b0, acc[4]);
    acc[5] = WMMA_BF16(a1, b1, acc[5]);
    acc[6] = WMMA_BF16(a1, b2, acc[6]);
    acc[7] = WMMA_BF16(a1, b3, acc[7]);
  }
}

// x_proj = embeds_bf16 (BTxEMBED) @ W_word^T + enc_proj[b]  -> (BT x G4) f32
// block (32,4): 4 waves, each a 32x64 tile; block covers 32(M) x 256(N).
__global__ __launch_bounds__(128) void xproj_gemm(
    const __bf16* __restrict__ A, const __bf16* __restrict__ W,
    const float* __restrict__ enc_proj, float* __restrict__ xproj) {
  const int lane = threadIdx.x;
  const int mbase = blockIdx.y * 32;
  const int nbase = blockIdx.x * 256 + threadIdx.y * 64;
  v8f z = {};
  v8f acc[8] = {z, z, z, z, z, z, z, z};
  mma_32x64(A + (size_t)mbase * EMBED, EMBED,
            W + (size_t)nbase * EMBED, EMBED, EMBED, lane, acc);
  const int half = lane >> 4, col = lane & 15;
#pragma unroll
  for (int mi = 0; mi < 2; ++mi)
#pragma unroll
    for (int ni = 0; ni < 4; ++ni) {
      v8f c = acc[mi * 4 + ni];
      int n = nbase + ni * 16 + col;
#pragma unroll
      for (int r = 0; r < 8; ++r) {
        int m = mbase + mi * 16 + half * 8 + r;
        int bat = m >> 5;
        xproj[(size_t)m * G4 + n] = c[r] + enc_proj[bat * G4 + n];
      }
    }
}

// z = x_proj[:, t, :] + h_bf (BBxHIDDEN) @ W_hh^T   -> (BB x G4) f32
__global__ __launch_bounds__(128) void lstm_gemm(
    const __bf16* __restrict__ hbf, const __bf16* __restrict__ Whh,
    const float* __restrict__ xproj, int t, float* __restrict__ z_out) {
  const int lane = threadIdx.x;
  const int mbase = blockIdx.y * 32;                   // batch base
  const int nbase = blockIdx.x * 256 + threadIdx.y * 64;
  v8f z = {};
  v8f acc[8] = {z, z, z, z, z, z, z, z};
  mma_32x64(hbf + (size_t)mbase * HIDDEN, HIDDEN,
            Whh + (size_t)nbase * HIDDEN, HIDDEN, HIDDEN, lane, acc);
  const int half = lane >> 4, col = lane & 15;
#pragma unroll
  for (int mi = 0; mi < 2; ++mi)
#pragma unroll
    for (int ni = 0; ni < 4; ++ni) {
      v8f c = acc[mi * 4 + ni];
      int n = nbase + ni * 16 + col;
#pragma unroll
      for (int r = 0; r < 8; ++r) {
        int m = mbase + mi * 16 + half * 8 + r;        // batch index
        z_out[(size_t)m * G4 + n] = c[r] + xproj[((size_t)m * TT + t) * G4 + n];
      }
    }
}

__global__ void lstm_act(const float* __restrict__ z,
                         float* __restrict__ h_cur, float* __restrict__ c_cur,
                         __bf16* __restrict__ h_bf, __bf16* __restrict__ hs_bf,
                         int t) {
  int idx = blockIdx.x * blockDim.x + threadIdx.x;     // BB*HIDDEN
  if (idx >= BB * HIDDEN) return;
  int b = idx >> 10;
  int j = idx & (HIDDEN - 1);
  const float* zb = z + (size_t)b * G4;
  float ig = zb[j];
  float fg = zb[HIDDEN + j];
  float gg = zb[2 * HIDDEN + j];
  float og = zb[3 * HIDDEN + j];
  float c = sigf(fg) * c_cur[idx] + sigf(ig) * tanhf(gg);
  float h = sigf(og) * tanhf(c);
  c_cur[idx] = c;
  h_cur[idx] = h;
  __bf16 hb = (__bf16)h;
  h_bf[idx] = hb;
  hs_bf[((size_t)b * TT + t) * HIDDEN + j] = hb;
}

// ---------------------------------------------------------------------------
// Dominant GEMM: logits = hs (BTxHIDDEN) @ fc_W^T + fc_b -> (BT x VOCAB).
// Workgroup (32,4): M tile 32, N tile 256 (4 waves x 64N).  The shared A-tile
// is staged into LDS by the Tensor Data Mover in 64-wide K chunks, double
// buffered (TENSORcnt + barriers); A frags then come from LDS (ds_load_b128),
// B streams from global (L2-resident).
// ---------------------------------------------------------------------------
__global__ __launch_bounds__(128) void logits_gemm(
    const __bf16* __restrict__ hs, const __bf16* __restrict__ fcW,
    const float* __restrict__ fcb, float* __restrict__ out) {
  __shared__ __align__(16) __bf16 ldsA[2][32 * 64];    // 8 KB double buffer
  const int lane = threadIdx.x;
  const int wv   = threadIdx.y;                        // 0..3
  const int mbase = blockIdx.y * 32;
  const int nbase = blockIdx.x * 256 + wv * 64;
  const __bf16* Atile = hs + (size_t)mbase * HIDDEN;
  const __bf16* Bbase = fcW + (size_t)nbase * HIDDEN;

  v8f z = {};
  v8f acc[8] = {z, z, z, z, z, z, z, z};

  // preload K-chunk 0
  if (wv == 0) {
    tdm_load_tile_2d((unsigned)(size_t)&ldsA[0][0], Atile, HIDDEN, 64, 32);
    __builtin_amdgcn_s_wait_tensorcnt(0);
  }
  __syncthreads();

  int cur = 0;
  const int NCHUNK = HIDDEN / 64;                      // 16
  for (int kc = 0; kc < NCHUNK; ++kc) {
    if (wv == 0 && kc + 1 < NCHUNK) {
      tdm_load_tile_2d((unsigned)(size_t)&ldsA[cur ^ 1][0],
                       Atile + (size_t)(kc + 1) * 64, HIDDEN, 64, 32);
    }
    const __bf16* As = &ldsA[cur][0];
#pragma unroll
    for (int kk = 0; kk < 64; kk += 32) {
      const int k0 = kc * 64 + kk;
      v16bf a0 = load_frag(As,           64, kk, lane);    // LDS
      v16bf a1 = load_frag(As + 16 * 64, 64, kk, lane);    // LDS
      v16bf b0 = load_frag(Bbase,               HIDDEN, k0, lane);
      v16bf b1 = load_frag(Bbase + 16 * HIDDEN, HIDDEN, k0, lane);
      v16bf b2 = load_frag(Bbase + 32 * HIDDEN, HIDDEN, k0, lane);
      v16bf b3 = load_frag(Bbase + 48 * HIDDEN, HIDDEN, k0, lane);
      acc[0] = WMMA_BF16(a0, b0, acc[0]);
      acc[1] = WMMA_BF16(a0, b1, acc[1]);
      acc[2] = WMMA_BF16(a0, b2, acc[2]);
      acc[3] = WMMA_BF16(a0, b3, acc[3]);
      acc[4] = WMMA_BF16(a1, b0, acc[4]);
      acc[5] = WMMA_BF16(a1, b1, acc[5]);
      acc[6] = WMMA_BF16(a1, b2, acc[6]);
      acc[7] = WMMA_BF16(a1, b3, acc[7]);
    }
    __syncthreads();                                   // done reading ldsA[cur]
    if (wv == 0) __builtin_amdgcn_s_wait_tensorcnt(0); // next chunk landed
    __syncthreads();                                   // publish to all waves
    cur ^= 1;
  }

  const int half = lane >> 4, col = lane & 15;
#pragma unroll
  for (int mi = 0; mi < 2; ++mi)
#pragma unroll
    for (int ni = 0; ni < 4; ++ni) {
      v8f c = acc[mi * 4 + ni];
      int n = nbase + ni * 16 + col;
      float bias = fcb[n];
#pragma unroll
      for (int r = 0; r < 8; ++r) {
        int m = mbase + mi * 16 + half * 8 + r;
        out[(size_t)m * VOCAB + n] = c[r] + bias;
      }
    }
}

__global__ void write_hn_cn(const float* __restrict__ h_cur,
                            const float* __restrict__ c_cur,
                            float* __restrict__ out) {
  int i = blockIdx.x * blockDim.x + threadIdx.x;       // BB*HIDDEN
  if (i >= BB * HIDDEN) return;
  out[LOGITS_ELEMS + i]               = h_cur[i];
  out[LOGITS_ELEMS + BB * HIDDEN + i] = c_cur[i];
}

// ---------------------------------------------------------------------------

extern "C" void kernel_launch(void* const* d_in, const int* in_sizes, int n_in,
                              void* d_out, int out_size, void* d_ws, size_t ws_size,
                              hipStream_t stream) {
  const int*   tokens = (const int*)  d_in[0];
  const float* enc    = (const float*)d_in[1];
  const float* h0     = (const float*)d_in[2];
  const float* c0     = (const float*)d_in[3];
  const float* table  = (const float*)d_in[4];
  const float* W_ih   = (const float*)d_in[5];
  const float* W_hh   = (const float*)d_in[6];
  const float* b_ih   = (const float*)d_in[7];
  const float* b_hh   = (const float*)d_in[8];
  const float* fc_W   = (const float*)d_in[9];
  const float* fc_b   = (const float*)d_in[10];
  float* out = (float*)d_out;

  char* ws = (char*)d_ws;
  size_t off = 0;
  auto carve = [&](size_t bytes) -> char* {
    char* p = ws + off;
    off = (off + bytes + 255) & ~(size_t)255;
    return p;
  };
  __bf16* wword_bf = (__bf16*)carve((size_t)G4 * EMBED * 2);
  __bf16* whh_bf   = (__bf16*)carve((size_t)G4 * HIDDEN * 2);
  __bf16* fcw_bf   = (__bf16*)carve((size_t)VOCAB * HIDDEN * 2);
  __bf16* emb_bf   = (__bf16*)carve((size_t)BT * EMBED * 2);
  float*  encp     = (float*) carve((size_t)BB * G4 * 4);
  float*  xproj    = (float*) carve((size_t)BT * G4 * 4);
  float*  zbuf     = (float*) carve((size_t)BB * G4 * 4);
  float*  h_cur    = (float*) carve((size_t)BB * HIDDEN * 4);
  float*  c_cur    = (float*) carve((size_t)BB * HIDDEN * 4);
  __bf16* h_bf     = (__bf16*)carve((size_t)BB * HIDDEN * 2);
  __bf16* hs_bf    = (__bf16*)carve((size_t)BT * HIDDEN * 2);
  (void)ws_size; (void)in_sizes; (void)n_in; (void)out_size;

  const int TPB = 256;

  cvt_wword<<<(G4 * EMBED + TPB - 1) / TPB, TPB, 0, stream>>>(W_ih, wword_bf);
  cvt_bf16_contig<<<(G4 * HIDDEN + TPB - 1) / TPB, TPB, 0, stream>>>(W_hh, whh_bf, G4 * HIDDEN);
  cvt_bf16_contig<<<(VOCAB * HIDDEN + TPB - 1) / TPB, TPB, 0, stream>>>(fc_W, fcw_bf, VOCAB * HIDDEN);
  embed_gather<<<(BT * EMBED + TPB - 1) / TPB, TPB, 0, stream>>>(tokens, table, emb_bf);
  enc_proj_kernel<<<(BB * G4 + TPB - 1) / TPB, TPB, 0, stream>>>(enc, W_ih, b_ih, b_hh, encp);
  init_state<<<(BB * HIDDEN + TPB - 1) / TPB, TPB, 0, stream>>>(h0, c0, h_cur, c_cur, h_bf);

  // x_proj GEMM: grid covers (BT x G4) in 32x256 blocks
  {
    dim3 blk(32, 4);
    dim3 grd(G4 / 256, BT / 32);          // (16, 64)
    xproj_gemm<<<grd, blk, 0, stream>>>(emb_bf, wword_bf, encp, xproj);
  }

  // sequential LSTM recurrence
  for (int t = 0; t < TT; ++t) {
    dim3 blk(32, 4);
    dim3 grd(G4 / 256, BB / 32);          // (16, 2)
    lstm_gemm<<<grd, blk, 0, stream>>>(h_bf, whh_bf, xproj, t, zbuf);
    lstm_act<<<(BB * HIDDEN + TPB - 1) / TPB, TPB, 0, stream>>>(zbuf, h_cur, c_cur, h_bf, hs_bf, t);
  }

  // dominant GEMM with TDM-staged A tile
  {
    dim3 blk(32, 4);
    dim3 grd(VOCAB / 256, BT / 32);       // (125, 64)
    logits_gemm<<<grd, blk, 0, stream>>>(hs_bf, fcw_bf, fc_b, out);
  }

  write_hn_cn<<<(BB * HIDDEN + TPB - 1) / TPB, TPB, 0, stream>>>(h_cur, c_cur, out);
}
